// BeamSearch_20048907338250
// MI455X (gfx1250) — matrix operand order repaired and checked by reading, not compiled, not run on hardware
//
#include <hip/hip_runtime.h>
#include <hip/hip_bf16.h>

// ---------------- problem constants (from reference) ----------------
#define BATCH   8
#define BEAM    32
#define DDIM    256
#define VOCAB   50257
#define VPAD    50432          // 197 * 256, multiple of 256
#define NT      (VPAD / 16)    // 3152 n-tiles of 16
#define MAXLEN  17             // 1 initial + 16 decoded
#define STEPS   16
#define MROWS   (BATCH * BEAM) // 256 rows in the big GEMM

typedef __attribute__((ext_vector_type(16))) __bf16 v16bf;
typedef __attribute__((ext_vector_type(8)))  float  v8f;

static __device__ __forceinline__ unsigned short f32_to_bf16(float f) {
    union { float f; unsigned u; } x; x.f = f;
    unsigned r = x.u + 0x7FFFu + ((x.u >> 16) & 1u);   // round-to-nearest-even
    return (unsigned short)(r >> 16);
}

// ---------------------------------------------------------------------------
// K0: convert W (f32, [D][V] row-major) into bf16 B-fragments, pre-swizzled to
// a contiguous-k 32x16 B-fragment lane layout (same k order as the A packing):
//   lane 0..15  : N = lane,     K = j       (j = 0..15)
//   lane 16..31 : N = lane-16,  K = 16 + j
// Fragment array: wb[((ntile*8 + kchunk)*32 + lane)*16 + j]
// ---------------------------------------------------------------------------
__global__ void conv_w_kernel(const float* __restrict__ W,
                              unsigned short* __restrict__ wb) {
    int ntile = blockIdx.x;          // 0..NT-1
    int c     = blockIdx.y;          // k-chunk 0..7
    int tid   = threadIdx.x;         // 0..511
    int lane  = tid >> 4;
    int j     = tid & 15;
    int n = ntile * 16 + (lane & 15);
    int h = lane >> 4;
    int k = c * 32 + h * 16 + j;
    float v = (n < VOCAB) ? W[(size_t)k * VOCAB + n] : 0.0f;
    wb[((size_t)(ntile * 8 + c) * 32 + lane) * 16 + j] = f32_to_bf16(v);
}

// ---------------------------------------------------------------------------
// K_init: cache rows <- state, ids[...][0] <- initial token, scores <- 0
// ---------------------------------------------------------------------------
__global__ void init_state_kernel(const int* __restrict__ initial_ids,
                                  const float* __restrict__ state,
                                  int* __restrict__ ids,
                                  float* __restrict__ scores,
                                  float* __restrict__ cache) {
    int row = blockIdx.x;            // 0..255 = b*32+k
    int b   = row >> 5;
    int e   = threadIdx.x;           // 0..255
    cache[row * DDIM + e] = state[b * DDIM + e];
    if (e == 0) { ids[row * MAXLEN] = initial_ids[b]; scores[row] = 0.0f; }
}

// ---------------------------------------------------------------------------
// K1: x = tanh(cache @ U + emb[last]); write f32 copy (new cache candidate)
// and bf16 A-fragments pre-swizzled to a contiguous-k 16x32 A-fragment layout (same k order as W's B packing):
//   lane 0..15  (h=0): M = lane, K = 0..15
//   lane 16..31 (h=1): M = lane-16, K = 16..31
// K -> (h, j):  h = K>>4,  j = K&15
// ---------------------------------------------------------------------------
__global__ void state_update_kernel(const float* __restrict__ cache,
                                    const int* __restrict__ ids_cur,
                                    int cur_len, int beam0,
                                    const float* __restrict__ emb,
                                    const float* __restrict__ U,
                                    float* __restrict__ x_f32,
                                    unsigned short* __restrict__ xa) {
    int row = blockIdx.x;                    // 0..255
    int b = row >> 5, k = row & 31;
    int kk = (k < beam0) ? k : 0;            // clamp: rows >= beam0 mirror row 0
    int rrow = b * BEAM + kk;
    int e = threadIdx.x;                     // output feature 0..255

    __shared__ float cs[DDIM];
    cs[e] = cache[rrow * DDIM + e];
    __syncthreads();

    float acc = 0.0f;
    #pragma unroll 8
    for (int d = 0; d < DDIM; ++d)
        acc = fmaf(cs[d], U[d * DDIM + e], acc);

    int last = ids_cur[rrow * MAXLEN + (cur_len - 1)];
    float xv = tanhf(acc + emb[(size_t)last * DDIM + e]);
    x_f32[row * DDIM + e] = xv;

    int rtile = row >> 4, ml = row & 15;
    int c   = e >> 5;                        // k-chunk
    int kin = e & 31;                        // K within chunk
    int h = kin >> 4;
    int j = kin & 15;
    int lane = h * 16 + ml;
    xa[((size_t)(rtile * 8 + c) * 32 + lane) * 16 + j] = f32_to_bf16(xv);
}

// ---------------------------------------------------------------------------
// K2: logits = x @ W via v_wmma_f32_16x16x32_bf16.
// Block = 256 threads (8 waves). Block tile: 64 rows x 256 cols.
//   wave w owns TWO row-tiles (A frags held in 128 VGPRs) and 4 n-tiles:
//     rtiles = blockIdx.y*4 + (w>>2)*2 + {0,1}
//     ntiles = blockIdx.x*16 + (w&3)*4 + {0..3}
// Every B fragment is loaded once and feeds two WMMAs (halves B L2 traffic
// vs. one row-tile per wave and gives the scheduler two independent WMMAs
// of latency cover per load clause). 64 WMMAs per wave.
// ---------------------------------------------------------------------------
__global__ void __launch_bounds__(256)
gemm_logits_kernel(const unsigned short* __restrict__ xa,
                   const unsigned short* __restrict__ wb,
                   float* __restrict__ logits) {
    int lane = threadIdx.x & 31;
    int wave = threadIdx.x >> 5;                        // 0..7
    int r0   = blockIdx.y * 4 + (wave >> 2) * 2;        // first rtile of pair
    int ngrp = blockIdx.x * 16 + (wave & 3) * 4;        // first of 4 n-tiles

    // load the 16 A-fragments (two row-tiles) once; reused by all 4 n-tiles
    const v16bf* ap0 = (const v16bf*)xa + ((size_t)r0 * 8) * 32 + lane;
    v16bf a0[8], a1[8];
    #pragma unroll
    for (int c = 0; c < 8; ++c) { a0[c] = ap0[c * 32]; a1[c] = ap0[256 + c * 32]; }

    for (int t = 0; t < 4; ++t) {
        int ntile = ngrp + t;
        const v16bf* bp = (const v16bf*)wb + ((size_t)ntile * 8) * 32 + lane;
        if (t < 3) __builtin_prefetch((const void*)(bp + 256), 0, 1); // next tile
        v8f acc0 = {};
        v8f acc1 = {};
        #pragma unroll
        for (int c = 0; c < 8; ++c) {
            v16bf bfr = bp[c * 32];
            acc0 = __builtin_amdgcn_wmma_f32_16x16x32_bf16(
                       false, a0[c], false, bfr, (short)0, acc0, false, false);
            acc1 = __builtin_amdgcn_wmma_f32_16x16x32_bf16(
                       false, a1[c], false, bfr, (short)0, acc1, false, false);
        }
        // C layout: lanes 0..15 -> N=lane, M=i ; lanes 16..31 -> N=lane-16, M=8+i
        int mb = (lane >> 4) << 3;
        int n  = ntile * 16 + (lane & 15);
        float* out0 = logits + (size_t)(r0 * 16 + mb) * VPAD + n;
        float* out1 = logits + (size_t)((r0 + 1) * 16 + mb) * VPAD + n;
        #pragma unroll
        for (int i = 0; i < 8; ++i) out0[(size_t)i * VPAD] = acc0[i];
        #pragma unroll
        for (int i = 0; i < 8; ++i) out1[(size_t)i * VPAD] = acc1[i];
    }
}

// ---------------------------------------------------------------------------
// K3: per-batch logsumexp + exact top-32 of (scores[k] + logit - lse[k]).
// One 1024-thread block per batch. Histogram-threshold + compaction + 32
// iterative argmaxes (descending order, matching jax.lax.top_k).
// ---------------------------------------------------------------------------
__global__ void __launch_bounds__(1024)
topk_kernel(const float* __restrict__ logits,
            const float* __restrict__ scores_cur,
            int beam0,
            float* __restrict__ top_scores,
            unsigned* __restrict__ top_ix) {
    int b   = blockIdx.x;
    int tid = threadIdx.x;
    int lane = tid & 31, w = tid >> 5;

    __shared__ float    addc[BEAM];      // scores[k] - lse[k]
    __shared__ float    rowmax[BEAM];
    __shared__ unsigned hist[4096];
    __shared__ float    cv[2048];
    __shared__ unsigned ci[2048];
    __shared__ float    rv[1024];
    __shared__ int      ri[1024];
    __shared__ unsigned cnt;
    __shared__ int      Tbin;
    __shared__ float    lo_s;

    // ---- Phase A: per-row online logsumexp (one wave per beam row) ----
    if (w < beam0) {
        const float* Lrow = logits + (size_t)(b * BEAM + w) * VPAD;
        float m = -1e30f, s = 0.0f;
        for (int v = lane; v < VOCAB; v += 32) {
            float x = Lrow[v];
            if (x > m) { s = s * __expf(m - x) + 1.0f; m = x; }
            else       { s += __expf(x - m); }
        }
        for (int off = 16; off > 0; off >>= 1) {
            float om = __shfl_xor(m, off, 32);
            float os = __shfl_xor(s, off, 32);
            if (om > m) { s = s * __expf(m - om) + os; m = om; }
            else        { s += os * __expf(om - m); }
        }
        if (lane == 0) {
            float lse = m + __logf(s);
            float sc  = scores_cur[b * BEAM + w];
            addc[w]   = sc - lse;
            rowmax[w] = sc - lse + m;
        }
    }
    for (int i = tid; i < 4096; i += 1024) hist[i] = 0;
    __syncthreads();
    if (tid == 0) {
        float am = -1e30f;
        for (int k2 = 0; k2 < beam0; ++k2) am = fmaxf(am, rowmax[k2]);
        lo_s = am - 32.0f;                 // 4096 bins over 32.0 range
        cnt = 0;
    }
    __syncthreads();
    float lo = lo_s;

    // ---- Phase B: histogram of adjusted scores ----
    for (int k2 = 0; k2 < beam0; ++k2) {
        const float ac = addc[k2];
        const float* Lrow = logits + (size_t)(b * BEAM + k2) * VPAD;
        for (int v = tid; v < VOCAB; v += 1024) {
            float a = Lrow[v] + ac;
            int bin = (int)((a - lo) * 128.0f);
            bin = bin < 0 ? 0 : (bin > 4095 ? 4095 : bin);
            atomicAdd(&hist[bin], 1u);
        }
    }
    __syncthreads();
    if (tid == 0) {
        unsigned cum = 0; int T = 0;
        for (int i = 4095; i >= 0; --i) { cum += hist[i]; if (cum >= 32u) { T = i; break; } }
        Tbin = T;
    }
    __syncthreads();
    int T = Tbin;

    // ---- Phase C: compact survivors into LDS ----
    for (int k2 = 0; k2 < beam0; ++k2) {
        const float ac = addc[k2];
        const float* Lrow = logits + (size_t)(b * BEAM + k2) * VPAD;
        for (int v = tid; v < VOCAB; v += 1024) {
            float a = Lrow[v] + ac;
            int bin = (int)((a - lo) * 128.0f);
            bin = bin < 0 ? 0 : (bin > 4095 ? 4095 : bin);
            if (bin >= T) {
                unsigned pos = atomicAdd(&cnt, 1u);
                if (pos < 2048u) { cv[pos] = a; ci[pos] = (unsigned)(k2 * VOCAB + v); }
            }
        }
    }
    __syncthreads();
    int n = (cnt < 2048u) ? (int)cnt : 2048;

    // ---- Phase D: 32 iterative argmaxes (descending) ----
    for (int j = 0; j < 32; ++j) {
        float bv = -1e30f; int bp = -1;
        for (int i = tid; i < n; i += 1024)
            if (cv[i] > bv) { bv = cv[i]; bp = i; }
        rv[tid] = bv; ri[tid] = bp;
        __syncthreads();
        for (int s2 = 512; s2 > 0; s2 >>= 1) {
            if (tid < s2 && rv[tid + s2] > rv[tid]) { rv[tid] = rv[tid + s2]; ri[tid] = ri[tid + s2]; }
            __syncthreads();
        }
        if (tid == 0) {
            int p = ri[0];
            top_scores[b * BEAM + j] = rv[0];
            top_ix[b * BEAM + j]     = (p >= 0) ? ci[p] : 0u;
            if (p >= 0) cv[p] = -1e30f;
        }
        __syncthreads();
    }
}

// ---------------------------------------------------------------------------
// K4: gather winning beams -> new ids / cache / scores (double-buffered)
// ---------------------------------------------------------------------------
__global__ void gather_update_kernel(const int* __restrict__ ids_cur,
                                     int* __restrict__ ids_next,
                                     const float* __restrict__ x_f32,
                                     float* __restrict__ cache_next,
                                     const float* __restrict__ top_scores,
                                     const unsigned* __restrict__ top_ix,
                                     float* __restrict__ scores_next,
                                     int cur_len) {
    int j = blockIdx.x, b = blockIdx.y;
    int t = threadIdx.x;                       // 0..255
    unsigned ix = top_ix[b * BEAM + j];
    int beam = (int)(ix / (unsigned)VOCAB);
    int tok  = (int)(ix % (unsigned)VOCAB);
    cache_next[(size_t)(b * BEAM + j) * DDIM + t] =
        x_f32[(size_t)(b * BEAM + beam) * DDIM + t];
    if (t < cur_len)
        ids_next[(b * BEAM + j) * MAXLEN + t] = ids_cur[(b * BEAM + beam) * MAXLEN + t];
    if (t == 0) {
        ids_next[(b * BEAM + j) * MAXLEN + cur_len] = tok;
        scores_next[b * BEAM + j] = top_scores[b * BEAM + j];
    }
}

// ---------------------------------------------------------------------------
// K5: emit (ids as float, scores) concatenated flat
// ---------------------------------------------------------------------------
__global__ void finalize_kernel(const int* __restrict__ ids,
                                const float* __restrict__ scores,
                                float* __restrict__ out) {
    int t = blockIdx.x * 256 + threadIdx.x;
    const int NID = BATCH * BEAM * MAXLEN;     // 4352
    if (t < NID)                 out[t] = (float)ids[t];
    else if (t < NID + BATCH*BEAM) out[t] = scores[t - NID];
}

// ---------------------------------------------------------------------------
extern "C" void kernel_launch(void* const* d_in, const int* in_sizes, int n_in,
                              void* d_out, int out_size, void* d_ws, size_t ws_size,
                              hipStream_t stream) {
    const int*   initial_ids = (const int*)  d_in[0];
    const float* state       = (const float*)d_in[1];
    const float* emb         = (const float*)d_in[2];
    const float* W           = (const float*)d_in[3];
    const float* U           = (const float*)d_in[4];
    // beam_size (d_in[5]) == 32, max_decode_length (d_in[6]) == 16 per reference.

    char* ws = (char*)d_ws;
    size_t off = 0;
    auto alloc = [&](size_t bytes) -> void* {
        off = (off + 255) & ~(size_t)255;
        void* p = ws + off;
        off += bytes;
        return p;
    };
    unsigned short* wb      = (unsigned short*)alloc((size_t)NT * 8 * 32 * 16 * sizeof(unsigned short));
    float*          logits  = (float*)         alloc((size_t)MROWS * VPAD * sizeof(float));
    unsigned short* xa      = (unsigned short*)alloc((size_t)16 * 8 * 32 * 16 * sizeof(unsigned short));
    float*          x_f32   = (float*)         alloc((size_t)MROWS * DDIM * sizeof(float));
    float*          cacheB[2] = { (float*)alloc((size_t)MROWS * DDIM * sizeof(float)),
                                  (float*)alloc((size_t)MROWS * DDIM * sizeof(float)) };
    int*            idsB[2]   = { (int*)alloc((size_t)MROWS * MAXLEN * sizeof(int)),
                                  (int*)alloc((size_t)MROWS * MAXLEN * sizeof(int)) };
    float*          scoresB[2]= { (float*)alloc(MROWS * sizeof(float)),
                                  (float*)alloc(MROWS * sizeof(float)) };
    float*          tops    = (float*)   alloc(MROWS * sizeof(float));
    unsigned*       topi    = (unsigned*)alloc(MROWS * sizeof(unsigned));

    conv_w_kernel<<<dim3(NT, 8), 512, 0, stream>>>(W, wb);
    init_state_kernel<<<MROWS, DDIM, 0, stream>>>(initial_ids, state,
                                                  idsB[0], scoresB[0], cacheB[0]);

    int cur = 0;
    for (int step = 0; step < STEPS; ++step) {
        int beam0   = (step == 0) ? 1 : BEAM;
        int cur_len = step + 1;
        state_update_kernel<<<MROWS, DDIM, 0, stream>>>(
            cacheB[cur], idsB[cur], cur_len, beam0, emb, U, x_f32, xa);
        gemm_logits_kernel<<<dim3(VPAD / 256, MROWS / 64), 256, 0, stream>>>(
            xa, wb, logits);
        topk_kernel<<<BATCH, 1024, 0, stream>>>(
            logits, scoresB[cur], beam0, tops, topi);
        gather_update_kernel<<<dim3(BEAM, BATCH), DDIM, 0, stream>>>(
            idsB[cur], idsB[1 - cur], x_f32, cacheB[1 - cur],
            tops, topi, scoresB[1 - cur], cur_len);
        cur ^= 1;
    }
    finalize_kernel<<<(BATCH * BEAM * MAXLEN + BATCH * BEAM + 255) / 256, 256, 0, stream>>>(
        idsB[cur], scoresB[cur], (float*)d_out);
}
